// SarathiAtten_8959301779832
// MI455X (gfx1250) — matrix operand order repaired and checked by reading, not compiled, not run on hardware
//
#include <hip/hip_runtime.h>
#include <hip/hip_bf16.h>

// SarathiAtten on MI455X (gfx1250): f16 WMMA pipeline
//   cast -> QKV GEMM (scatter Q*SCALE, K, V^T) -> flash attention -> out GEMM
// GEMMs use 128x256 block tiles, 64x64 wave tiles (4x4 WMMA frags) to halve
// LDS traffic per v_wmma (0.5 KB/WMMA) so the XDL pipe, not ds, is the limit.

typedef __attribute__((ext_vector_type(16))) _Float16 v16h;
typedef __attribute__((ext_vector_type(8)))  _Float16 v8h;
typedef __attribute__((ext_vector_type(8)))  float    v8f;

#define SEQ    2048
#define BATCH  2
#define HID    2048
#define NHEAD  16
#define HDIM   128
#define MROWS  (SEQ*BATCH)   // 4096
#define QKV_N  (3*HID)       // 6144
#define SCALE_F (1.0f/11.313708498984761f)

// ---------------- WMMA fragment helpers (ISA 7.12.2 layouts) ----------------

// A-matrix 16x32 f16: lane<16 -> row=lane, K {0..7,16..23}; lane>=16 -> row=lane-16, K {8..15,24..31}
__device__ __forceinline__ v16h load_a_frag(const _Float16* base, int stride) {
  const int lane = threadIdx.x & 31;
  const int hi   = lane >> 4;
  const int r    = lane & 15;
  const _Float16* p = base + (size_t)r * stride + hi * 8;
  v8h lo = *(const v8h*)(p);
  v8h hh = *(const v8h*)(p + 16);
  v16h f;
#pragma unroll
  for (int i = 0; i < 8; ++i) { f[i] = lo[i]; f[i + 8] = hh[i]; }
  return f;
}

// B-matrix 32x16 f16: lane<16 -> col=lane, K {0..7,16..23}; lane>=16 -> col=lane-16, K {8..15,24..31}
// Source is row-major [N][K] (i.e. column n of B is a contiguous row of the source).
__device__ __forceinline__ v16h load_b_frag(const _Float16* base, int stride) {
  const int lane = threadIdx.x & 31;
  const int hi   = lane >> 4;
  const int n    = lane & 15;
  const _Float16* p = base + (size_t)n * stride + hi * 8;
  v8h lo = *(const v8h*)(p);
  v8h hh = *(const v8h*)(p + 16);
  v16h f;
#pragma unroll
  for (int i = 0; i < 8; ++i) { f[i] = lo[i]; f[i + 8] = hh[i]; }
  return f;
}

__device__ __forceinline__ v8f wmma_f16(v16h a, v16h b, v8f c) {
  return __builtin_amdgcn_wmma_f32_16x16x32_f16(false, a, false, b, (short)0, c, false, false);
}

// ---------------- stage 0: fp32 -> f16 cast ----------------

__global__ __launch_bounds__(256) void cast_f32_to_f16(const float* __restrict__ src,
                                                       _Float16* __restrict__ dst, int n8) {
  int i = blockIdx.x * blockDim.x + threadIdx.x;
  if (i >= n8) return;
  const float4* s = (const float4*)src + (size_t)i * 2;
  float4 a = s[0], b = s[1];
  v8h o;
  o[0] = (_Float16)a.x; o[1] = (_Float16)a.y; o[2] = (_Float16)a.z; o[3] = (_Float16)a.w;
  o[4] = (_Float16)b.x; o[5] = (_Float16)b.y; o[6] = (_Float16)b.z; o[7] = (_Float16)b.w;
  *((v8h*)dst + i) = o;
}

// ---------------- GEMM tiling ----------------
// Block: 256 threads = 8 waves in a 2(M) x 4(N) grid; wave tile 64x64 = 4x4 frags.
#define BM  128
#define BN  256
#define BK  64
#define LDK (BK + 8)   // padded LDS stride (row pitch 144B: 16B-aligned, bank-rotating)

// ---------------- stage 1: QKV GEMM with Q/K/V^T scatter epilogue ----------------

__global__ __launch_bounds__(256) void qkv_gemm(const _Float16* __restrict__ X,
                                                const _Float16* __restrict__ W,
                                                _Float16* __restrict__ Qh,
                                                _Float16* __restrict__ Kh,
                                                _Float16* __restrict__ Vt) {
  __shared__ alignas(16) _Float16 As[BM][LDK];
  __shared__ alignas(16) _Float16 Bs[BN][LDK];
  const int tid  = threadIdx.x;
  const int lane = tid & 31, wv = tid >> 5;
  const int hi = lane >> 4, ln = lane & 15;
  const int wm = (wv & 1) * 64, wn = (wv >> 1) * 64;     // 2x4 wave grid, 64x64 tiles
  const int m0 = blockIdx.y * BM, n0 = blockIdx.x * BN;
  v8f acc[4][4] = {};
  const int arow = tid >> 1, aoff = (tid & 1) * 32;
  for (int k0 = 0; k0 < HID; k0 += BK) {
    const v8h* ga = (const v8h*)(X + (size_t)(m0 + arow) * HID + k0 + aoff);
#pragma unroll
    for (int c = 0; c < 4; ++c) *(v8h*)&As[arow][aoff + c * 8] = ga[c];
    const v8h* gb = (const v8h*)(W + (size_t)(n0 + tid) * HID + k0);
#pragma unroll
    for (int c = 0; c < 8; ++c) *(v8h*)&Bs[tid][c * 8] = gb[c];
    __syncthreads();
#pragma unroll
    for (int kk = 0; kk < BK; kk += 32) {
      v16h af[4], bf[4];
#pragma unroll
      for (int i = 0; i < 4; ++i) af[i] = load_a_frag(&As[wm + i * 16][kk], LDK);
#pragma unroll
      for (int j = 0; j < 4; ++j) bf[j] = load_b_frag(&Bs[wn + j * 16][kk], LDK);
#pragma unroll
      for (int i = 0; i < 4; ++i)
#pragma unroll
        for (int j = 0; j < 4; ++j) acc[i][j] = wmma_f16(af[i], bf[j], acc[i][j]);
    }
    __syncthreads();
  }
  // Scatter: col = head*384 + c; c<128 -> Q (scaled), c<256 -> K, else -> V transposed.
#pragma unroll
  for (int i = 0; i < 4; ++i)
#pragma unroll
    for (int j = 0; j < 4; ++j) {
      const int nn = n0 + wn + j * 16 + ln;
      const int head = nn / 384, c = nn % 384;
#pragma unroll
      for (int r = 0; r < 8; ++r) {
        const int mm = m0 + wm + i * 16 + r + 8 * hi;
        const int s = mm >> 1, b = mm & 1;         // row = s*BATCH + b
        const int bh = b * NHEAD + head;
        const float v = acc[i][j][r];
        if (c < HDIM)
          Qh[((size_t)bh * SEQ + s) * HDIM + c] = (_Float16)(v * SCALE_F);
        else if (c < 2 * HDIM)
          Kh[((size_t)bh * SEQ + s) * HDIM + (c - HDIM)] = (_Float16)v;
        else
          Vt[((size_t)bh * HDIM + (c - 2 * HDIM)) * SEQ + s] = (_Float16)v;
      }
    }
}

// ---------------- stage 2: flash attention (one wave per 16 query rows) ----------------

__global__ __launch_bounds__(128) void flash_attn(const _Float16* __restrict__ Qh,
                                                  const _Float16* __restrict__ Kh,
                                                  const _Float16* __restrict__ Vt,
                                                  _Float16* __restrict__ Ch) {
  __shared__ alignas(16) _Float16 plds[4][16][40];  // wave-private P tiles, padded
  const int tid  = threadIdx.x;
  const int lane = tid & 31, wv = tid >> 5;
  const int hi = lane >> 4, ln = lane & 15;
  const int bh = blockIdx.x >> 5;                   // (b, head) pair, 0..31
  const int q0 = (blockIdx.x & 31) * 64 + wv * 16;  // query row base for this wave
  const int b = bh >> 4, head = bh & 15;
  const _Float16* Qg = Qh + (size_t)bh * SEQ * HDIM;
  const _Float16* Kg = Kh + (size_t)bh * SEQ * HDIM;
  const _Float16* Vg = Vt + (size_t)bh * HDIM * SEQ;

  v16h qf[4];
#pragma unroll
  for (int kq = 0; kq < 4; ++kq)
    qf[kq] = load_a_frag(Qg + (size_t)q0 * HDIM + kq * 32, HDIM);

  v8f o[8] = {};
  float mr[8], lr[8];
#pragma unroll
  for (int r = 0; r < 8; ++r) { mr[r] = -1e30f; lr[r] = 0.0f; }

  for (int t0 = 0; t0 < SEQ; t0 += 32) {
    // S = Q * K^T for keys [t0, t0+32)
    v8f s0 = {}, s1 = {};
#pragma unroll
    for (int kq = 0; kq < 4; ++kq) {
      v16h kf0 = load_b_frag(Kg + (size_t)t0 * HDIM + kq * 32, HDIM);
      v16h kf1 = load_b_frag(Kg + (size_t)(t0 + 16) * HDIM + kq * 32, HDIM);
      s0 = wmma_f16(qf[kq], kf0, s0);
      s1 = wmma_f16(qf[kq], kf1, s1);
    }
    // online softmax: this lane's slot r is row (r + 8*hi) in both S and O frags
    float alpha[8];
#pragma unroll
    for (int r = 0; r < 8; ++r) {
      float mx = fmaxf(s0[r], s1[r]);
#pragma unroll
      for (int off = 1; off < 16; off <<= 1) mx = fmaxf(mx, __shfl_xor(mx, off, 16));
      const float mn = fmaxf(mr[r], mx);
      alpha[r] = __expf(mr[r] - mn);
      const float p0 = __expf(s0[r] - mn);
      const float p1 = __expf(s1[r] - mn);
      float ps = p0 + p1;
#pragma unroll
      for (int off = 1; off < 16; off <<= 1) ps += __shfl_xor(ps, off, 16);
      lr[r] = lr[r] * alpha[r] + ps;
      mr[r] = mn;
      plds[wv][r + 8 * hi][ln]      = (_Float16)p0;   // D-layout -> row-major P
      plds[wv][r + 8 * hi][16 + ln] = (_Float16)p1;
    }
    asm volatile("s_wait_dscnt 0x0" ::: "memory");    // P stores -> P A-frag reload
    v16h pf = load_a_frag(&plds[wv][0][0], 40);
    // O = O*alpha + P(16x32) * V(32x128); Vt rows are contiguous along t
#pragma unroll
    for (int h = 0; h < 8; ++h) {
      v8f t = o[h];
#pragma unroll
      for (int r = 0; r < 8; ++r) t[r] *= alpha[r];
      v16h vf = load_b_frag(Vg + (size_t)(h * 16) * SEQ + t0, SEQ);
      o[h] = wmma_f16(pf, vf, t);
    }
  }
  // normalize and store ctx in (s, b, head*128 + d) layout for the output GEMM
#pragma unroll
  for (int h = 0; h < 8; ++h)
#pragma unroll
    for (int r = 0; r < 8; ++r) {
      const int s = q0 + r + 8 * hi;
      const float v = o[h][r] / lr[r];
      Ch[((size_t)s * BATCH + b) * HID + head * HDIM + h * 16 + ln] = (_Float16)v;
    }
}

// ---------------- stage 3: output projection GEMM (f32 out) ----------------

__global__ __launch_bounds__(256) void out_gemm(const _Float16* __restrict__ X,
                                                const _Float16* __restrict__ W,
                                                float* __restrict__ out) {
  __shared__ alignas(16) _Float16 As[BM][LDK];
  __shared__ alignas(16) _Float16 Bs[BN][LDK];
  const int tid  = threadIdx.x;
  const int lane = tid & 31, wv = tid >> 5;
  const int hi = lane >> 4, ln = lane & 15;
  const int wm = (wv & 1) * 64, wn = (wv >> 1) * 64;
  const int m0 = blockIdx.y * BM, n0 = blockIdx.x * BN;
  v8f acc[4][4] = {};
  const int arow = tid >> 1, aoff = (tid & 1) * 32;
  for (int k0 = 0; k0 < HID; k0 += BK) {
    const v8h* ga = (const v8h*)(X + (size_t)(m0 + arow) * HID + k0 + aoff);
#pragma unroll
    for (int c = 0; c < 4; ++c) *(v8h*)&As[arow][aoff + c * 8] = ga[c];
    const v8h* gb = (const v8h*)(W + (size_t)(n0 + tid) * HID + k0);
#pragma unroll
    for (int c = 0; c < 8; ++c) *(v8h*)&Bs[tid][c * 8] = gb[c];
    __syncthreads();
#pragma unroll
    for (int kk = 0; kk < BK; kk += 32) {
      v16h af[4], bf[4];
#pragma unroll
      for (int i = 0; i < 4; ++i) af[i] = load_a_frag(&As[wm + i * 16][kk], LDK);
#pragma unroll
      for (int j = 0; j < 4; ++j) bf[j] = load_b_frag(&Bs[wn + j * 16][kk], LDK);
#pragma unroll
      for (int i = 0; i < 4; ++i)
#pragma unroll
        for (int j = 0; j < 4; ++j) acc[i][j] = wmma_f16(af[i], bf[j], acc[i][j]);
    }
    __syncthreads();
  }
#pragma unroll
  for (int i = 0; i < 4; ++i)
#pragma unroll
    for (int j = 0; j < 4; ++j) {
      const int nn = n0 + wn + j * 16 + ln;
#pragma unroll
      for (int r = 0; r < 8; ++r) {
        const int mm = m0 + wm + i * 16 + r + 8 * hi;
        out[(size_t)mm * HID + nn] = acc[i][j][r];
      }
    }
}

// ---------------- launcher ----------------

extern "C" void kernel_launch(void* const* d_in, const int* in_sizes, int n_in,
                              void* d_out, int out_size, void* d_ws, size_t ws_size,
                              hipStream_t stream) {
  const float* hs    = (const float*)d_in[0];  // (S, B, H) f32
  const float* w_qkv = (const float*)d_in[1];  // (3H, H) f32
  const float* w_out = (const float*)d_in[2];  // (H, H) f32
  float* out = (float*)d_out;                  // (S, B, H) f32

  char* ws = (char*)d_ws;                      // ~117.5 MB of f16 scratch
  const size_t szXh = (size_t)MROWS * HID * sizeof(_Float16);
  const size_t szWh = (size_t)QKV_N * HID * sizeof(_Float16);
  const size_t szWo = (size_t)HID * HID * sizeof(_Float16);
  const size_t szHd = (size_t)BATCH * NHEAD * SEQ * HDIM * sizeof(_Float16);
  _Float16* Xh  = (_Float16*)(ws);
  _Float16* Wh  = (_Float16*)(ws + szXh);
  _Float16* Woh = (_Float16*)(ws + szXh + szWh);
  _Float16* Qh  = (_Float16*)(ws + szXh + szWh + szWo);
  _Float16* Kh  = (_Float16*)(ws + szXh + szWh + szWo + szHd);
  _Float16* Vt  = (_Float16*)(ws + szXh + szWh + szWo + 2 * szHd);
  _Float16* Ch  = (_Float16*)(ws + szXh + szWh + szWo + 3 * szHd);

  int n8;
  n8 = (int)((size_t)MROWS * HID / 8);
  cast_f32_to_f16<<<(n8 + 255) / 256, 256, 0, stream>>>(hs, Xh, n8);
  n8 = (int)((size_t)QKV_N * HID / 8);
  cast_f32_to_f16<<<(n8 + 255) / 256, 256, 0, stream>>>(w_qkv, Wh, n8);
  n8 = (int)((size_t)HID * HID / 8);
  cast_f32_to_f16<<<(n8 + 255) / 256, 256, 0, stream>>>(w_out, Woh, n8);

  qkv_gemm<<<dim3(QKV_N / BN, MROWS / BM), 256, 0, stream>>>(Xh, Wh, Qh, Kh, Vt);
  flash_attn<<<dim3(BATCH * NHEAD * (SEQ / 64)), 128, 0, stream>>>(Qh, Kh, Vt, Ch);
  out_gemm<<<dim3(HID / BN, MROWS / BM), 256, 0, stream>>>(Ch, Woh, out);
}
